// Encoder_26053271617788
// MI455X (gfx1250) — matrix operand order repaired and checked by reading, not compiled, not run on hardware
//
#include <hip/hip_runtime.h>
#include <stddef.h>

#define N_NODES 100000
#define N_EDGES 3200000
#define D_IN    512
#define D_HID   256
#define D_LAT   128

typedef __attribute__((ext_vector_type(16))) __bf16 v16bf;
typedef __attribute__((ext_vector_type(8)))  float  v8f;

union BFrag {
    v16bf v;
    unsigned short s[16];
    uint4 q[2];
};

__device__ __forceinline__ unsigned short f2bf(float x) {
    // round-to-nearest-even float32 -> bf16 (bits)
    unsigned u = __float_as_uint(x);
    unsigned r = u + 0x7FFFu + ((u >> 16) & 1u);
    return (unsigned short)(r >> 16);
}

// -------------------------------------------------------------------------
// Elementwise f32 -> bf16 (8 elements / thread, 16B stores)
// -------------------------------------------------------------------------
__global__ __launch_bounds__(256) void cvt_f32_to_bf16(const float* __restrict__ src,
                                                       unsigned short* __restrict__ dst,
                                                       long long n) {
    long long i = ((long long)blockIdx.x * blockDim.x + threadIdx.x) * 8;
    if (i >= n) return;
    float4 f0 = *(const float4*)(src + i);
    float4 f1 = *(const float4*)(src + i + 4);
    union { uint4 u; unsigned short s[8]; } o;
    o.s[0] = f2bf(f0.x); o.s[1] = f2bf(f0.y); o.s[2] = f2bf(f0.z); o.s[3] = f2bf(f0.w);
    o.s[4] = f2bf(f1.x); o.s[5] = f2bf(f1.y); o.s[6] = f2bf(f1.z); o.s[7] = f2bf(f1.w);
    *(uint4*)(dst + i) = o.u;
}

// -------------------------------------------------------------------------
// h_bf16 = bf16(relu(hacc + b1))   (8 elements / thread)
// -------------------------------------------------------------------------
__global__ __launch_bounds__(256) void relu_bias_to_bf16(const float* __restrict__ hacc,
                                                         const float* __restrict__ b1,
                                                         unsigned short* __restrict__ hbf) {
    size_t i = ((size_t)blockIdx.x * blockDim.x + threadIdx.x) * 8;
    if (i >= (size_t)N_NODES * D_HID) return;
    const int cb = (int)(i & (D_HID - 1));
    float4 f0 = *(const float4*)(hacc + i);
    float4 f1 = *(const float4*)(hacc + i + 4);
    float4 b0 = *(const float4*)(b1 + cb);
    float4 b1v = *(const float4*)(b1 + cb + 4);
    float e0 = fmaxf(f0.x + b0.x, 0.f), e1 = fmaxf(f0.y + b0.y, 0.f);
    float e2 = fmaxf(f0.z + b0.z, 0.f), e3 = fmaxf(f0.w + b0.w, 0.f);
    float e4 = fmaxf(f1.x + b1v.x, 0.f), e5 = fmaxf(f1.y + b1v.y, 0.f);
    float e6 = fmaxf(f1.z + b1v.z, 0.f), e7 = fmaxf(f1.w + b1v.w, 0.f);
    union { uint4 u; unsigned short s[8]; } o;
    o.s[0] = f2bf(e0); o.s[1] = f2bf(e1); o.s[2] = f2bf(e2); o.s[3] = f2bf(e3);
    o.s[4] = f2bf(e4); o.s[5] = f2bf(e5); o.s[6] = f2bf(e6); o.s[7] = f2bf(e7);
    *(uint4*)(hbf + i) = o.u;
}

// -------------------------------------------------------------------------
// Pack a [K x N] f32 weight matrix into WMMA B-fragment lane order, bf16.
// B-fragment layout (16-bit B, 32xK x 16): lane half = lane>>4 covers
// K = half*8 + e (e<8), 16 + half*8 + (e-8) (e>=8), column = ntile*16 + (lane & 15).
// -------------------------------------------------------------------------
__global__ void pack_w_bf16(const float* __restrict__ W, unsigned short* __restrict__ Wpk,
                            int K, int N) {
    const int ksteps = K >> 5;
    const int ntiles = N >> 4;
    const int total  = ksteps * ntiles * 32;
    int t = blockIdx.x * blockDim.x + threadIdx.x;
    if (t >= total) return;
    const int lane = t & 31;
    const int tmp  = t >> 5;
    const int nt   = tmp % ntiles;
    const int kst  = tmp / ntiles;
    const int half = lane >> 4;
    const int ncol = nt * 16 + (lane & 15);
    const int kbase = kst * 32 + half * 8;
    unsigned short* dst = Wpk + (size_t)t * 16;
#pragma unroll
    for (int e = 0; e < 16; ++e)
        dst[e] = f2bf(W[(size_t)(kbase + (e < 8 ? e : 8 + e)) * N + ncol]);
}

// -------------------------------------------------------------------------
// GEMM1: XW1[100000,256] = Xbf[100000,512] @ W1[512,256]  (bf16 WMMA, f32 acc)
// grid (6250, 2), block 256 = 8 waves; wave -> one 16x16 N-tile.
// A fragment = two 16B loads per k-step (K runs half*8+e and 16+half*8+e).
// -------------------------------------------------------------------------
__global__ __launch_bounds__(256) void gemm_xw1(const unsigned short* __restrict__ Xbf,
                                                const unsigned short* __restrict__ W1pk,
                                                float* __restrict__ Y) {
    const int lane  = threadIdx.x & 31;
    const int wave  = threadIdx.x >> 5;
    const int mtile = blockIdx.x;
    const int ntile = blockIdx.y * 8 + wave;      // 0..15
    const int l16   = lane & 15;
    const int half  = lane >> 4;

    const unsigned short* xrow = Xbf + (size_t)(mtile * 16 + l16) * D_IN;
    v8f acc = {};

#pragma unroll 4
    for (int ks = 0; ks < D_IN / 32; ++ks) {
        BFrag a;
        a.q[0] = *(const uint4*)(xrow + ks * 32 + half * 8);
        a.q[1] = *(const uint4*)(xrow + ks * 32 + 16 + half * 8);

        BFrag b;
        b.v = *(const v16bf*)(W1pk + (((size_t)ks * 16 + ntile) * 32 + lane) * 16);

        acc = __builtin_amdgcn_wmma_f32_16x16x32_bf16(
                  false, a.v, false, b.v, (short)0, acc, false, false);
    }

    float* yb = Y + (size_t)(mtile * 16 + half * 8) * D_HID + ntile * 16 + l16;
#pragma unroll
    for (int r = 0; r < 8; ++r)
        yb[(size_t)r * D_HID] = acc[r];
}

// -------------------------------------------------------------------------
// GEMM2: HW2 = h @ W2, HW3 = h @ W3 (shared A fragment per k-step)
// grid (6250), block 256 = 8 waves; wave -> N-tile of the 128-wide outputs.
// -------------------------------------------------------------------------
__global__ __launch_bounds__(256) void gemm_h_w23(const unsigned short* __restrict__ Hbf,
                                                  const unsigned short* __restrict__ W2pk,
                                                  const unsigned short* __restrict__ W3pk,
                                                  float* __restrict__ HW2,
                                                  float* __restrict__ HW3) {
    const int lane  = threadIdx.x & 31;
    const int ntile = threadIdx.x >> 5;           // 0..7
    const int mtile = blockIdx.x;
    const int l16   = lane & 15;
    const int half  = lane >> 4;

    const unsigned short* hrow = Hbf + (size_t)(mtile * 16 + l16) * D_HID;
    v8f acc2 = {};
    v8f acc3 = {};

#pragma unroll
    for (int ks = 0; ks < D_HID / 32; ++ks) {
        BFrag a;
        a.q[0] = *(const uint4*)(hrow + ks * 32 + half * 8);
        a.q[1] = *(const uint4*)(hrow + ks * 32 + 16 + half * 8);

        const size_t bidx = (((size_t)ks * 8 + ntile) * 32 + lane) * 16;
        BFrag b2;  b2.v = *(const v16bf*)(W2pk + bidx);
        BFrag b3;  b3.v = *(const v16bf*)(W3pk + bidx);

        acc2 = __builtin_amdgcn_wmma_f32_16x16x32_bf16(
                   false, a.v, false, b2.v, (short)0, acc2, false, false);
        acc3 = __builtin_amdgcn_wmma_f32_16x16x32_bf16(
                   false, a.v, false, b3.v, (short)0, acc3, false, false);
    }

    const size_t off = (size_t)(mtile * 16 + half * 8) * D_LAT + ntile * 16 + l16;
#pragma unroll
    for (int r = 0; r < 8; ++r) {
        HW2[off + (size_t)r * D_LAT] = acc2[r];
        HW3[off + (size_t)r * D_LAT] = acc3[r];
    }
}

// -------------------------------------------------------------------------
// SpMM via per-edge wave-parallel f32 atomics; gather sources are L2-resident.
// -------------------------------------------------------------------------
__global__ __launch_bounds__(256) void spmm_acc256(const int* __restrict__ row,
                                                   const int* __restrict__ col,
                                                   const float* __restrict__ vals,
                                                   const float* __restrict__ src,
                                                   float* __restrict__ dst) {
    const int lane   = threadIdx.x & 31;
    const int warp   = blockIdx.x * (blockDim.x >> 5) + (threadIdx.x >> 5);
    const int nwarps = gridDim.x * (blockDim.x >> 5);
    for (int e = warp; e < N_EDGES; e += nwarps) {
        if (lane == 0) {
            __builtin_prefetch(row + e + nwarps, 0, 1);
            __builtin_prefetch(col + e + nwarps, 0, 1);
            __builtin_prefetch(vals + e + nwarps, 0, 1);
        }
        const int   r = row[e];
        const int   c = col[e];
        const float v = vals[e];
        const float* s = src + (size_t)c * D_HID;
        float*       d = dst + (size_t)r * D_HID;
#pragma unroll
        for (int j = 0; j < D_HID / 32; ++j) {
            float x = v * s[lane + 32 * j];
            __hip_atomic_fetch_add(d + lane + 32 * j, x,
                                   __ATOMIC_RELAXED, __HIP_MEMORY_SCOPE_AGENT);
        }
    }
}

__global__ __launch_bounds__(256) void spmm_acc128x2(const int* __restrict__ row,
                                                     const int* __restrict__ col,
                                                     const float* __restrict__ vals,
                                                     const float* __restrict__ src2,
                                                     const float* __restrict__ src3,
                                                     float* __restrict__ dst2,
                                                     float* __restrict__ dst3) {
    const int lane   = threadIdx.x & 31;
    const int warp   = blockIdx.x * (blockDim.x >> 5) + (threadIdx.x >> 5);
    const int nwarps = gridDim.x * (blockDim.x >> 5);
    for (int e = warp; e < N_EDGES; e += nwarps) {
        if (lane == 0) {
            __builtin_prefetch(row + e + nwarps, 0, 1);
            __builtin_prefetch(col + e + nwarps, 0, 1);
            __builtin_prefetch(vals + e + nwarps, 0, 1);
        }
        const int   r = row[e];
        const int   c = col[e];
        const float v = vals[e];
        const float* s2 = src2 + (size_t)c * D_LAT;
        const float* s3 = src3 + (size_t)c * D_LAT;
        float*       d2 = dst2 + (size_t)r * D_LAT;
        float*       d3 = dst3 + (size_t)r * D_LAT;
#pragma unroll
        for (int j = 0; j < D_LAT / 32; ++j) {
            const int cc = lane + 32 * j;
            __hip_atomic_fetch_add(d2 + cc, v * s2[cc],
                                   __ATOMIC_RELAXED, __HIP_MEMORY_SCOPE_AGENT);
            __hip_atomic_fetch_add(d3 + cc, v * s3[cc],
                                   __ATOMIC_RELAXED, __HIP_MEMORY_SCOPE_AGENT);
        }
    }
}

__global__ __launch_bounds__(256) void add_bias_out(float* __restrict__ out,
                                                    const float* __restrict__ b2,
                                                    const float* __restrict__ b3) {
    const size_t half = (size_t)N_NODES * D_LAT;
    size_t i = (size_t)blockIdx.x * blockDim.x + threadIdx.x;
    if (i >= 2 * half) return;
    const float b = (i < half) ? b2[i & (D_LAT - 1)] : b3[i & (D_LAT - 1)];
    out[i] += b;
}

// -------------------------------------------------------------------------
extern "C" void kernel_launch(void* const* d_in, const int* in_sizes, int n_in,
                              void* d_out, int out_size, void* d_ws, size_t ws_size,
                              hipStream_t stream) {
    const float* X     = (const float*)d_in[0];
    const int*   erow  = (const int*)  d_in[1];
    const int*   ecol  = (const int*)  d_in[2];
    const float* evals = (const float*)d_in[3];
    const float* W1    = (const float*)d_in[4];
    const float* b1    = (const float*)d_in[5];
    const float* W2    = (const float*)d_in[6];
    const float* b2    = (const float*)d_in[7];
    const float* W3    = (const float*)d_in[8];
    const float* b3    = (const float*)d_in[9];
    float* outp = (float*)d_out;

    // Workspace layout with phase-based aliasing (all offsets 32B-aligned):
    //  region A (102.4MB): xbf (phase 1-2), then hbf (phase 4-5)
    //  region B (102.4MB): xw1 (phase 2-3), then hw2+hw3 (phase 5-6)
    //  region C (102.4MB): hacc (phase 3-4)
    //  region D: packed bf16 weights
    char* ws = (char*)d_ws;
    const size_t SZ_NH = (size_t)N_NODES * D_HID * sizeof(float);   // 102.4 MB
    const size_t SZ_NL = (size_t)N_NODES * D_LAT * sizeof(float);   //  51.2 MB
    unsigned short* xbf  = (unsigned short*)(ws);                   // region A
    unsigned short* hbf  = (unsigned short*)(ws);                   // region A (after xbf dead)
    float*          xw1  = (float*)(ws + SZ_NH);                    // region B
    float*          hw2  = (float*)(ws + SZ_NH);                    // region B (after xw1 dead)
    float*          hw3  = (float*)(ws + SZ_NH + SZ_NL);            // region B second half
    float*          hacc = (float*)(ws + 2 * SZ_NH);                // region C
    unsigned short* w1pk = (unsigned short*)(ws + 3 * SZ_NH);       // region D
    unsigned short* w2pk = w1pk + (size_t)D_IN  * D_HID;
    unsigned short* w3pk = w2pk + (size_t)D_HID * D_LAT;

    // zero atomic accumulators (graph-capture-safe memset nodes)
    hipMemsetAsync(hacc,  0, SZ_NH,     stream);
    hipMemsetAsync(d_out, 0, 2 * SZ_NL, stream);

    // pack weights into bf16 WMMA B-fragment layout
    pack_w_bf16<<<(16 * 16 * 32 + 255) / 256, 256, 0, stream>>>(W1, w1pk, D_IN,  D_HID);
    pack_w_bf16<<<( 8 *  8 * 32 + 255) / 256, 256, 0, stream>>>(W2, w2pk, D_HID, D_LAT);
    pack_w_bf16<<<( 8 *  8 * 32 + 255) / 256, 256, 0, stream>>>(W3, w3pk, D_HID, D_LAT);

    // phase 1: X -> bf16
    const long long NX = (long long)N_NODES * D_IN;
    cvt_f32_to_bf16<<<(unsigned)((NX / 8 + 255) / 256), 256, 0, stream>>>(X, xbf, NX);

    // phase 2: GEMM1
    const int MTILES = N_NODES / 16;   // 6250, exact
    gemm_xw1<<<dim3(MTILES, 2), 256, 0, stream>>>(xbf, w1pk, xw1);

    // phase 3: SpMM1 (atomic segment-sum)
    spmm_acc256<<<2048, 256, 0, stream>>>(erow, ecol, evals, xw1, hacc);

    // phase 4: relu + bias, emit bf16 A-operand for GEMM2 (overwrites xbf region)
    const size_t NH = (size_t)N_NODES * D_HID;
    relu_bias_to_bf16<<<(unsigned)((NH / 8 + 255) / 256), 256, 0, stream>>>(hacc, b1, hbf);

    // phase 5: GEMM2 (dual B) -> hw2/hw3 (overwrites xw1 region)
    gemm_h_w23<<<MTILES, 256, 0, stream>>>(hbf, w2pk, w3pk, hw2, hw3);

    // phase 6: SpMM2 straight into d_out halves, then bias
    spmm_acc128x2<<<2048, 256, 0, stream>>>(erow, ecol, evals, hw2, hw3,
                                            outp, outp + (size_t)N_NODES * D_LAT);
    add_bias_out<<<(unsigned)((2 * (size_t)N_NODES * D_LAT + 255) / 256), 256, 0, stream>>>(outp, b2, b3);
}
